// get_model_6932077216015
// MI455X (gfx1250) — compile-verified, matrix operand order checked by reading, not run on hardware
//
#include <hip/hip_runtime.h>

typedef __attribute__((ext_vector_type(16))) _Float16 v16h;
typedef __attribute__((ext_vector_type(8)))  _Float16 v8h;
typedef __attribute__((ext_vector_type(8)))  float    v8f;

#define BATCH 16
#define NPTS  2048

__global__ void split_kernel(const float* __restrict__ xyz,
                             float* __restrict__ coords, float* __restrict__ norm)
{
    int g = blockIdx.x * blockDim.x + threadIdx.x;
    int total = BATCH * 5 * NPTS;
    if (g >= total) return;
    int n  = g % NPTS;
    int t  = g / NPTS;
    int ch = t % 5;
    int b  = t / 5;
    float v = xyz[g];
    if (ch < 3) coords[((size_t)b * NPTS + n) * 3 + ch] = v;
    else        norm  [((size_t)b * NPTS + n) * 2 + (ch - 3)] = v;
}

__global__ void fps_kernel(const float* __restrict__ pts, int* __restrict__ out,
                           int n, int npoint)
{
    int b = blockIdx.x;
    const float* P = pts + (size_t)b * n * 3;
    __shared__ float dist[NPTS];
    __shared__ float rv[256];
    __shared__ int   ri[256];
    __shared__ int   sfar;
    int t = threadIdx.x;
    for (int i = t; i < n; i += blockDim.x) dist[i] = 1e10f;
    if (t == 0) sfar = 0;
    __syncthreads();
    for (int i = 0; i < npoint; ++i) {
        int far = sfar;
        if (t == 0) out[b * npoint + i] = far;
        float fx = P[far * 3 + 0], fy = P[far * 3 + 1], fz = P[far * 3 + 2];
        float bv = -1.f; int bi = 0;
        for (int j = t; j < n; j += blockDim.x) {
            float dx = P[j * 3 + 0] - fx;
            float dy = P[j * 3 + 1] - fy;
            float dz = P[j * 3 + 2] - fz;
            float d  = dx * dx + dy * dy + dz * dz;
            float nd = fminf(dist[j], d);
            dist[j] = nd;
            if (nd > bv) { bv = nd; bi = j; }
        }
        rv[t] = bv; ri[t] = bi;
        __syncthreads();
        for (int s = blockDim.x / 2; s > 0; s >>= 1) {
            if (t < s) {
                if (rv[t + s] > rv[t] || (rv[t + s] == rv[t] && ri[t + s] < ri[t])) {
                    rv[t] = rv[t + s]; ri[t] = ri[t + s];
                }
            }
            __syncthreads();
        }
        if (t == 0) sfar = ri[0];
        __syncthreads();
    }
}

__global__ void gather3_kernel(const float* __restrict__ coords, const int* __restrict__ fidx,
                               float* __restrict__ out, int Nprev, int S)
{
    int g = blockIdx.x * blockDim.x + threadIdx.x;
    int total = BATCH * S * 3;
    if (g >= total) return;
    int d = g % 3;
    int t = g / 3;
    int s = t % S;
    int b = t / S;
    int j = fidx[b * S + s];
    out[g] = coords[((size_t)b * Nprev + j) * 3 + d];
}

__global__ void ball_kernel(const float* __restrict__ xyz, const float* __restrict__ new_xyz,
                            int* __restrict__ idx, int Nsrc, int S, int K, float r2)
{
    int g = blockIdx.x * blockDim.x + threadIdx.x;
    if (g >= BATCH * S) return;
    int b = g / S;
    const float* P = xyz + (size_t)b * Nsrc * 3;
    float qx = new_xyz[(size_t)g * 3 + 0];
    float qy = new_xyz[(size_t)g * 3 + 1];
    float qz = new_xyz[(size_t)g * 3 + 2];
    int* o = idx + (size_t)g * K;
    int cnt = 0, first = 0;
    bool have = false;
    for (int j = 0; j < Nsrc && cnt < K; ++j) {
        float dx = P[j * 3 + 0] - qx;
        float dy = P[j * 3 + 1] - qy;
        float dz = P[j * 3 + 2] - qz;
        float d = dx * dx + dy * dy + dz * dz;
        if (d <= r2) {
            if (!have) { first = j; have = true; }
            o[cnt++] = j;
        }
    }
    for (; cnt < K; ++cnt) o[cnt] = first;
}

__global__ void group_kernel(const float* __restrict__ coords, const float* __restrict__ feats,
                             const int* __restrict__ idx, const float* __restrict__ new_xyz,
                             _Float16* __restrict__ Xh,
                             int Nprev, int S, int K, int Cf, int Cpad, int s0, int sc)
{
    size_t g = blockIdx.x * (size_t)blockDim.x + threadIdx.x;
    size_t total = (size_t)BATCH * sc * K * Cpad;
    if (g >= total) return;
    int c = (int)(g % Cpad);
    size_t row = g / Cpad;
    int k  = (int)(row % K);
    size_t t = row / K;
    int sl = (int)(t % sc);
    int b  = (int)(t / sc);
    int s  = s0 + sl;
    int j  = idx[((size_t)b * S + s) * K + k];
    float val;
    if (c < Cf) {
        val = feats[((size_t)b * Nprev + j) * Cf + c];
    } else if (c < Cf + 3) {
        int d = c - Cf;
        val = coords[((size_t)b * Nprev + j) * 3 + d] - new_xyz[((size_t)b * S + s) * 3 + d];
    } else {
        val = 0.f;
    }
    Xh[g] = (_Float16)val;
}

__global__ void sa3_concat_kernel(const float* __restrict__ l2x, const float* __restrict__ l2p,
                                  _Float16* __restrict__ Xh)
{
    int g = blockIdx.x * blockDim.x + threadIdx.x;
    int total = BATCH * 128 * 672;
    if (g >= total) return;
    int c = g % 672;
    int row = g / 672;   // b*128 + p
    float v;
    if (c < 3)        v = l2x[(size_t)row * 3 + c];
    else if (c < 643) v = l2p[(size_t)row * 640 + (c - 3)];
    else              v = 0.f;
    Xh[g] = (_Float16)v;
}

__global__ void prep_kernel(const float* __restrict__ w, const float* __restrict__ b,
                            const float* __restrict__ g, const float* __restrict__ bt,
                            const float* __restrict__ rm, const float* __restrict__ rv,
                            _Float16* __restrict__ Wh, float* __restrict__ scale,
                            float* __restrict__ shift,
                            int O, int C, int Opad, int Cpad, int has_bn)
{
    size_t i = blockIdx.x * (size_t)blockDim.x + threadIdx.x;
    size_t total = (size_t)Opad * Cpad;
    if (i >= total) return;
    int c = (int)(i % Cpad);
    int o = (int)(i / Cpad);
    float wv = (o < O && c < C) ? w[(size_t)o * C + c] : 0.f;
    Wh[i] = (_Float16)wv;
    if (c == 0) {
        float s = 0.f, t = 0.f;
        if (o < O) {
            if (has_bn) {
                s = g[o] * rsqrtf(rv[o] + 1e-5f);
                t = (b[o] - rm[o]) * s + bt[o];
            } else {
                s = 1.f;
                t = b[o];
            }
        }
        scale[o] = s; shift[o] = t;
    }
}

// ---------------------------------------------------------------------------
// Register-blocked WMMA GEMM: one wave computes a 32x32 output tile
// (2 M-subtiles x 2 N-subtiles, 4 accumulators). A/B fragments are each
// reused twice -> 2 b128 loads per v_wmma_f32_16x16x32_f16.
// The M-tail predicate (head GEMMs have M=16) is loop-invariant and hoisted:
// two straight-line K-loops so the compiler can clause loads and software-
// pipeline WMMAs with partial s_wait_loadcnt. EXEC is all-ones for every
// executed WMMA (all guards are wave-uniform).
//   X: (M x C) f16 row-major, W: (Opad x C) f16 row-major, C % 32 == 0
//   Y = epilogue( X * W^T ): per-channel scale/shift (+ReLU), f16/f32 stores
// ---------------------------------------------------------------------------
__global__ __launch_bounds__(128)
void gemm_wmma_kernel(const _Float16* __restrict__ X, const _Float16* __restrict__ W,
                      const float* __restrict__ scale, const float* __restrict__ shift,
                      _Float16* __restrict__ Yh, float* __restrict__ Yf,
                      int M, int C, int Opad, int Ostore, int ldYf, int relu)
{
    int tileN2 = blockIdx.y * blockDim.y + threadIdx.y;  // wave-uniform 32-col tile
    if (tileN2 * 32 >= Opad) return;                      // whole-wave exit
    int tileM2 = blockIdx.x;                              // 32-row tile
    int lane = threadIdx.x;
    int r  = lane & 15;
    int hi = lane >> 4;

    const bool hasM1 = (tileM2 * 32 + 16) < M;            // wave-uniform, loop-invariant

    const _Float16* xr0 = X + (size_t)(tileM2 * 32 + r) * C;
    const _Float16* xr1 = X + (size_t)(tileM2 * 32 + 16 + r) * C;
    const _Float16* wr0 = W + (size_t)(tileN2 * 32 + r) * C;
    const _Float16* wr1 = W + (size_t)(tileN2 * 32 + 16 + r) * C;

    v8f acc00 = {}, acc01 = {}, acc10 = {}, acc11 = {};
    const int klo = hi * 8;

    if (hasM1) {
        // common path: branch-free 8-load / 4-WMMA loop body
        for (int c0 = 0; c0 < C; c0 += 32) {
            union { v16h v; v8h h[2]; } a0, a1, b0, b1;
            a0.h[0] = *(const v8h*)(xr0 + c0 + klo);
            a0.h[1] = *(const v8h*)(xr0 + c0 + 16 + klo);
            a1.h[0] = *(const v8h*)(xr1 + c0 + klo);
            a1.h[1] = *(const v8h*)(xr1 + c0 + 16 + klo);
            b0.h[0] = *(const v8h*)(wr0 + c0 + klo);
            b0.h[1] = *(const v8h*)(wr0 + c0 + 16 + klo);
            b1.h[0] = *(const v8h*)(wr1 + c0 + klo);
            b1.h[1] = *(const v8h*)(wr1 + c0 + 16 + klo);
            acc00 = __builtin_amdgcn_wmma_f32_16x16x32_f16(
                false, a0.v, false, b0.v, (short)0, acc00, false, false);
            acc01 = __builtin_amdgcn_wmma_f32_16x16x32_f16(
                false, a0.v, false, b1.v, (short)0, acc01, false, false);
            acc10 = __builtin_amdgcn_wmma_f32_16x16x32_f16(
                false, a1.v, false, b0.v, (short)0, acc10, false, false);
            acc11 = __builtin_amdgcn_wmma_f32_16x16x32_f16(
                false, a1.v, false, b1.v, (short)0, acc11, false, false);
        }
    } else {
        // M-tail path (M == 16 head GEMMs): 6-load / 2-WMMA loop body
        for (int c0 = 0; c0 < C; c0 += 32) {
            union { v16h v; v8h h[2]; } a0, b0, b1;
            a0.h[0] = *(const v8h*)(xr0 + c0 + klo);
            a0.h[1] = *(const v8h*)(xr0 + c0 + 16 + klo);
            b0.h[0] = *(const v8h*)(wr0 + c0 + klo);
            b0.h[1] = *(const v8h*)(wr0 + c0 + 16 + klo);
            b1.h[0] = *(const v8h*)(wr1 + c0 + klo);
            b1.h[1] = *(const v8h*)(wr1 + c0 + 16 + klo);
            acc00 = __builtin_amdgcn_wmma_f32_16x16x32_f16(
                false, a0.v, false, b0.v, (short)0, acc00, false, false);
            acc01 = __builtin_amdgcn_wmma_f32_16x16x32_f16(
                false, a0.v, false, b1.v, (short)0, acc01, false, false);
        }
    }

    int col0 = tileN2 * 32 + r;
    int col1 = col0 + 16;
    float s0 = scale[col0], t0 = shift[col0];
    float s1 = scale[col1], t1 = shift[col1];
    int rowBase0 = tileM2 * 32 + hi * 8;
    int rowBase1 = rowBase0 + 16;

#pragma unroll
    for (int v = 0; v < 8; ++v) {
        float v00 = acc00[v] * s0 + t0;
        float v01 = acc01[v] * s1 + t1;
        if (relu) {
            v00 = v00 > 0.f ? v00 : 0.f;
            v01 = v01 > 0.f ? v01 : 0.f;
        }
        int row = rowBase0 + v;
        if (Yh) {
            Yh[(size_t)row * Opad + col0] = (_Float16)v00;
            Yh[(size_t)row * Opad + col1] = (_Float16)v01;
        }
        if (Yf) {
            if (col0 < Ostore) Yf[(size_t)row * ldYf + col0] = v00;
            if (col1 < Ostore) Yf[(size_t)row * ldYf + col1] = v01;
        }
    }
    if (hasM1) {
#pragma unroll
        for (int v = 0; v < 8; ++v) {
            float v10 = acc10[v] * s0 + t0;
            float v11 = acc11[v] * s1 + t1;
            if (relu) {
                v10 = v10 > 0.f ? v10 : 0.f;
                v11 = v11 > 0.f ? v11 : 0.f;
            }
            int row = rowBase1 + v;
            if (Yh) {
                Yh[(size_t)row * Opad + col0] = (_Float16)v10;
                Yh[(size_t)row * Opad + col1] = (_Float16)v11;
            }
            if (Yf) {
                if (col0 < Ostore) Yf[(size_t)row * ldYf + col0] = v10;
                if (col1 < Ostore) Yf[(size_t)row * ldYf + col1] = v11;
            }
        }
    }
}

__global__ void maxk_kernel(const float* __restrict__ Y, float* __restrict__ feats,
                            int K, int O, int sc, int s0, int Sfull, int Cstride, int c_off)
{
    size_t g = blockIdx.x * (size_t)blockDim.x + threadIdx.x;
    size_t total = (size_t)BATCH * sc * O;
    if (g >= total) return;
    int o = (int)(g % O);
    size_t t = g / O;
    int sl = (int)(t % sc);
    int b  = (int)(t / sc);
    const float* p = Y + ((size_t)(b * sc + sl) * K) * O + o;
    float m = p[0];
    for (int k = 1; k < K; ++k) m = fmaxf(m, p[(size_t)k * O]);
    feats[((size_t)b * Sfull + (s0 + sl)) * Cstride + c_off + o] = m;
}

__global__ void f2h_kernel(const float* __restrict__ in, _Float16* __restrict__ out, int n)
{
    int g = blockIdx.x * blockDim.x + threadIdx.x;
    if (g < n) out[g] = (_Float16)in[g];
}

struct LayerDesc {
    int wi, bi, gi, bti, rmi, rvi;
    int O, C, Opad, Cpad, has_bn;
    _Float16* Wh; float* sc; float* sh;
};

extern "C" void kernel_launch(void* const* d_in, const int* in_sizes, int n_in,
                              void* d_out, int out_size, void* d_ws, size_t ws_size,
                              hipStream_t stream)
{
    (void)in_sizes; (void)n_in; (void)out_size; (void)ws_size;
    char* ws = (char*)d_ws;
    size_t off = 0;
    auto alloc = [&](size_t bytes) -> char* {
        char* p = ws + off;
        off = (off + bytes + 255) & ~(size_t)255;
        return p;
    };
    auto F = [&](int i) -> const float* { return (const float*)d_in[i]; };

    // persistent buffers
    float* coords  = (float*)alloc((size_t)BATCH * NPTS * 3 * 4);
    float* normf   = (float*)alloc((size_t)BATCH * NPTS * 2 * 4);
    float* l1x     = (float*)alloc((size_t)BATCH * 512 * 3 * 4);
    float* l1p     = (float*)alloc((size_t)BATCH * 512 * 320 * 4);
    float* l2x     = (float*)alloc((size_t)BATCH * 128 * 3 * 4);
    float* l2p     = (float*)alloc((size_t)BATCH * 128 * 640 * 4);
    int*   fidx1   = (int*)alloc((size_t)BATCH * 512 * 4);
    int*   fidx2   = (int*)alloc((size_t)BATCH * 128 * 4);
    int*   ballidx = (int*)alloc((size_t)BATCH * 512 * 128 * 4);
    _Float16* sa3X = (_Float16*)alloc((size_t)BATCH * 128 * 672 * 2);
    _Float16* l3h  = (_Float16*)alloc((size_t)BATCH * 2048 * 2);
    // chunk ping-pong buffers (M_chunk max 65536 rows)
    _Float16* bufX  = (_Float16*)alloc((size_t)65536 * 352 * 2);
    _Float16* bufY1 = (_Float16*)alloc((size_t)65536 * 128 * 2);
    _Float16* bufY2 = (_Float16*)alloc((size_t)65536 * 128 * 2);
    float*    bufC  = (float*)alloc((size_t)65536 * 256 * 4);

    // Flat d_in order (dict insertion order at every level):
    //  0: xyz
    //  sa1: base 1,  (branch*3+layer)*6 + {w,b,g,bt,rm,rv}
    //  sa2: base 55,  sa3: base 109
    //  head: fc1.w=127 .b=128 | bn1 g,bt,rm,rv=129..132 | fc2.w=133 .b=134 |
    //        bn2=135..138 | fc3.w=139 .b=140
    LayerDesc L[24];
    int li = 0;
    auto addConv = [&](int base, int br, int ly, int O, int C, int Cpad) {
        int w = base + (br * 3 + ly) * 6;
        L[li] = { w, w + 1, w + 2, w + 3, w + 4, w + 5,
                  O, C, O, Cpad, 1, nullptr, nullptr, nullptr };
        ++li;
    };
    // SA1: L[0..8]
    addConv(1, 0, 0, 32, 5, 32);  addConv(1, 0, 1, 32, 32, 32);   addConv(1, 0, 2, 64, 32, 32);
    addConv(1, 1, 0, 64, 5, 32);  addConv(1, 1, 1, 64, 64, 64);   addConv(1, 1, 2, 128, 64, 64);
    addConv(1, 2, 0, 64, 5, 32);  addConv(1, 2, 1, 96, 64, 64);   addConv(1, 2, 2, 128, 96, 96);
    // SA2: L[9..17]
    addConv(55, 0, 0, 64, 323, 352);  addConv(55, 0, 1, 64, 64, 64);    addConv(55, 0, 2, 128, 64, 64);
    addConv(55, 1, 0, 128, 323, 352); addConv(55, 1, 1, 128, 128, 128); addConv(55, 1, 2, 256, 128, 128);
    addConv(55, 2, 0, 128, 323, 352); addConv(55, 2, 1, 128, 128, 128); addConv(55, 2, 2, 256, 128, 128);
    // SA3: L[18..20]
    addConv(109, 0, 0, 256, 643, 672);
    addConv(109, 0, 1, 512, 256, 256);
    addConv(109, 0, 2, 2048, 512, 512);
    // Head: L[21..23]
    L[21] = { 127, 128, 129, 130, 131, 132, 2048, 2048, 2048, 2048, 1, nullptr, nullptr, nullptr };
    L[22] = { 133, 134, 135, 136, 137, 138, 256, 2048, 256, 2048, 1, nullptr, nullptr, nullptr };
    L[23] = { 139, 140, -1, -1, -1, -1, 126, 256, 128, 256, 0, nullptr, nullptr, nullptr };

    for (int i = 0; i < 24; ++i) {
        LayerDesc& y = L[i];
        y.Wh = (_Float16*)alloc((size_t)y.Opad * y.Cpad * 2);
        y.sc = (float*)alloc((size_t)y.Opad * 4);
        y.sh = (float*)alloc((size_t)y.Opad * 4);
        size_t total = (size_t)y.Opad * y.Cpad;
        prep_kernel<<<dim3((unsigned)((total + 255) / 256)), dim3(256), 0, stream>>>(
            F(y.wi), F(y.bi),
            y.has_bn ? F(y.gi) : nullptr, y.has_bn ? F(y.bti) : nullptr,
            y.has_bn ? F(y.rmi) : nullptr, y.has_bn ? F(y.rvi) : nullptr,
            y.Wh, y.sc, y.sh, y.O, y.C, y.Opad, y.Cpad, y.has_bn);
    }

    auto gemm = [&](const _Float16* X, const LayerDesc& y, _Float16* Yh, float* Yf,
                    int M, int Ostore, int ldYf, int relu) {
        dim3 blk(32, 4);
        unsigned tilesN2 = (unsigned)(y.Opad / 32);
        if (tilesN2 == 0) tilesN2 = 1;
        dim3 grd((unsigned)((M + 31) / 32), (tilesN2 + 3) / 4);
        gemm_wmma_kernel<<<grd, blk, 0, stream>>>(
            X, y.Wh, y.sc, y.sh, Yh, Yf, M, y.Cpad, y.Opad, Ostore, ldYf, relu);
    };

    auto run_branch = [&](const float* coordsPrev, const float* featsPrev, int Nprev,
                          const float* newxyz, int S, int K, int sc_chunk, int Cf,
                          const LayerDesc* lys, float* featsOut, int Cstride, int c_off,
                          float radius) {
        int tot = BATCH * S;
        ball_kernel<<<dim3((tot + 255) / 256), dim3(256), 0, stream>>>(
            coordsPrev, newxyz, ballidx, Nprev, S, K, radius * radius);
        int Cpad0 = lys[0].Cpad;
        for (int s0 = 0; s0 < S; s0 += sc_chunk) {
            int M = BATCH * sc_chunk * K;
            size_t gt = (size_t)M * Cpad0;
            group_kernel<<<dim3((unsigned)((gt + 255) / 256)), dim3(256), 0, stream>>>(
                coordsPrev, featsPrev, ballidx, newxyz, bufX,
                Nprev, S, K, Cf, Cpad0, s0, sc_chunk);
            gemm(bufX,  lys[0], bufY1, nullptr, M, 0, 0, 1);
            gemm(bufY1, lys[1], bufY2, nullptr, M, 0, 0, 1);
            gemm(bufY2, lys[2], nullptr, bufC, M, lys[2].O, lys[2].O, 1);
            size_t mt = (size_t)BATCH * sc_chunk * lys[2].O;
            maxk_kernel<<<dim3((unsigned)((mt + 255) / 256)), dim3(256), 0, stream>>>(
                bufC, featsOut, K, lys[2].O, sc_chunk, s0, S, Cstride, c_off);
        }
    };

    // split
    {
        int tot = BATCH * 5 * NPTS;
        split_kernel<<<dim3((tot + 255) / 256), dim3(256), 0, stream>>>(F(0), coords, normf);
    }

    // SA1
    fps_kernel<<<dim3(BATCH), dim3(256), 0, stream>>>(coords, fidx1, NPTS, 512);
    gather3_kernel<<<dim3((BATCH * 512 * 3 + 255) / 256), dim3(256), 0, stream>>>(
        coords, fidx1, l1x, NPTS, 512);
    run_branch(coords, normf, NPTS, l1x, 512,  16, 256, 2, &L[0], l1p, 320,   0, 0.1f);
    run_branch(coords, normf, NPTS, l1x, 512,  32, 128, 2, &L[3], l1p, 320,  64, 0.2f);
    run_branch(coords, normf, NPTS, l1x, 512, 128,  32, 2, &L[6], l1p, 320, 192, 0.4f);

    // SA2
    fps_kernel<<<dim3(BATCH), dim3(256), 0, stream>>>(l1x, fidx2, 512, 128);
    gather3_kernel<<<dim3((BATCH * 128 * 3 + 255) / 256), dim3(256), 0, stream>>>(
        l1x, fidx2, l2x, 512, 128);
    run_branch(l1x, l1p, 512, l2x, 128,  32, 128, 320, &L[9],  l2p, 640,   0, 0.2f);
    run_branch(l1x, l1p, 512, l2x, 128,  64,  64, 320, &L[12], l2p, 640, 128, 0.4f);
    run_branch(l1x, l1p, 512, l2x, 128, 128,  32, 320, &L[15], l2p, 640, 384, 0.8f);

    // SA3 + global max-pool -> l3 (second model output)
    float* out_x  = (float*)d_out;                 // (16,126)
    float* out_l3 = out_x + (size_t)BATCH * 126;   // (16,2048)
    {
        int tot = BATCH * 128 * 672;
        sa3_concat_kernel<<<dim3((tot + 255) / 256), dim3(256), 0, stream>>>(l2x, l2p, sa3X);
        int M = BATCH * 128;   // 2048 rows
        gemm(sa3X,  L[18], bufY1, nullptr, M, 0, 0, 1);
        gemm(bufY1, L[19], bufY2, nullptr, M, 0, 0, 1);
        gemm(bufY2, L[20], nullptr, bufC, M, 2048, 2048, 1);
        size_t mt = (size_t)BATCH * 2048;
        maxk_kernel<<<dim3((unsigned)((mt + 255) / 256)), dim3(256), 0, stream>>>(
            bufC, out_l3, 128, 2048, 1, 0, 1, 2048, 0);
        f2h_kernel<<<dim3((BATCH * 2048 + 255) / 256), dim3(256), 0, stream>>>(
            out_l3, l3h, BATCH * 2048);
    }

    // Head: fc1+bn1+relu -> fc2+bn2+relu -> fc3 -> x (first model output)
    gemm(l3h,   L[21], bufY1, nullptr, BATCH, 0, 0, 1);
    gemm(bufY1, L[22], bufY2, nullptr, BATCH, 0, 0, 1);
    gemm(bufY2, L[23], nullptr, out_x, BATCH, 126, 126, 0);
}
